// TokenMixer_53446573031618
// MI455X (gfx1250) — compile-verified, matrix operand order checked by reading, not run on hardware
//
#include <hip/hip_runtime.h>
#include <hip/hip_bf16.h>

// ---------------------------------------------------------------------------
// Problem dims (fixed by the reference)
// ---------------------------------------------------------------------------
#define BB 4
#define TT 2048
#define DD 1024
#define HH 16
#define HD 64
#define MROWS (BB * TT)          // 8192

typedef __bf16 bf16x16 __attribute__((ext_vector_type(16)));
typedef __bf16 bf16x8  __attribute__((ext_vector_type(8)));
typedef __bf16 bf16x4  __attribute__((ext_vector_type(4)));
typedef float  v8f     __attribute__((ext_vector_type(8)));

// ---------------------------------------------------------------------------
// WMMA wrapper: D = A(16x32 bf16) * B(32x16 bf16) + C(16x16 f32)
// ---------------------------------------------------------------------------
__device__ __forceinline__ v8f wmma_bf16(bf16x16 a, bf16x16 b, v8f c) {
  return __builtin_amdgcn_wmma_f32_16x16x32_bf16(
      /*neg_a=*/false, a, /*neg_b=*/false, b,
      /*c_mod=*/(short)0, c, /*reuse_a=*/false, /*reuse_b=*/false);
}

// ---------------------------------------------------------------------------
// Load a 16x32 bf16 fragment (A-operand layout) from row-major memory.
// Lane L holds row = L%16; element e holds K = (e<8?0:16) + (L/16)*8 + (e%8).
// Both 8-element groups are contiguous -> two 16B loads.
// Same loader serves the B operand when memory is stored [N][K].
// ---------------------------------------------------------------------------
__device__ __forceinline__ bf16x16 load_frag_rm(const __bf16* base, int ld) {
  int lane = threadIdx.x & 31;
  int row  = lane & 15;
  int half = lane >> 4;
  const __bf16* p = base + (size_t)row * ld + half * 8;
  union { bf16x16 v; bf16x8 h[2]; } u;
  u.h[0] = *(const bf16x8*)(p);
  u.h[1] = *(const bf16x8*)(p + 16);
  return u.v;
}

// ---------------------------------------------------------------------------
// Kernel 1: LayerNorm (fp32 in) -> bf16 normalized activations
// ---------------------------------------------------------------------------
__global__ __launch_bounds__(256) void ln_kernel(const float* __restrict__ x,
                                                 const float* __restrict__ gamma,
                                                 const float* __restrict__ beta,
                                                 __bf16* __restrict__ xn) {
  int row = blockIdx.x;
  int tid = threadIdx.x;
  const float* xr = x + (size_t)row * DD;
  float4 v4 = ((const float4*)xr)[tid];
  float vv[4] = {v4.x, v4.y, v4.z, v4.w};
  float s = vv[0] + vv[1] + vv[2] + vv[3];
  float ss = vv[0]*vv[0] + vv[1]*vv[1] + vv[2]*vv[2] + vv[3]*vv[3];
#pragma unroll
  for (int m = 16; m >= 1; m >>= 1) {
    s  += __shfl_xor(s,  m, 32);
    ss += __shfl_xor(ss, m, 32);
  }
  __shared__ float ps[8], pss[8];
  int wave = tid >> 5, lane = tid & 31;
  if (lane == 0) { ps[wave] = s; pss[wave] = ss; }
  __syncthreads();
  if (wave == 0) {
    float a = (lane < 8) ? ps[lane]  : 0.f;
    float b = (lane < 8) ? pss[lane] : 0.f;
#pragma unroll
    for (int m = 4; m >= 1; m >>= 1) {
      a += __shfl_xor(a, m, 32);
      b += __shfl_xor(b, m, 32);
    }
    if (lane == 0) { ps[0] = a; pss[0] = b; }
  }
  __syncthreads();
  float mu  = ps[0]  * (1.0f / DD);
  float var = pss[0] * (1.0f / DD) - mu * mu;
  float rs  = rsqrtf(var + 1e-5f);
  __bf16* xo = xn + (size_t)row * DD;
  int c = tid * 4;
  bf16x4 y4;
#pragma unroll
  for (int j = 0; j < 4; ++j)
    y4[j] = (__bf16)((vv[j] - mu) * rs * gamma[c + j] + beta[c + j]);
  *(bf16x4*)(xo + c) = y4;
}

// ---------------------------------------------------------------------------
// Kernel 2: fused fp32 -> bf16 cast + transpose of weights.
// in:  [K][N] fp32 (row-major), out: [N][K] bf16 (row-major).
// 32x32 LDS-tiled so both global read and write are coalesced.
// ---------------------------------------------------------------------------
__global__ __launch_bounds__(256) void castT_kernel(const float* __restrict__ in,
                                                    __bf16* __restrict__ out,
                                                    int K, int N) {
  __shared__ float tile[32][33];
  int kt = blockIdx.y * 32;
  int nt = blockIdx.x * 32;
  int tx = threadIdx.x & 31;   // fastest dim
  int ty = threadIdx.x >> 5;   // 8 rows per pass
#pragma unroll
  for (int i = 0; i < 32; i += 8)
    tile[ty + i][tx] = in[(size_t)(kt + ty + i) * N + nt + tx];
  __syncthreads();
#pragma unroll
  for (int i = 0; i < 32; i += 8)
    out[(size_t)(nt + ty + i) * K + kt + tx] = (__bf16)tile[tx][ty + i];
}

// ---------------------------------------------------------------------------
// Shared GEMM mainloop: block tile 128(M) x 128(N), 8 waves.
// waveM = wave&3 (32 rows), waveN = wave>>2 (64 cols); each wave computes
// 2x4 = 8 WMMA subtiles per K-step from 6 LDS fragment loads.
// A row-major [M][K]; B supplied PRE-TRANSPOSED row-major [N][K], so both
// staging paths are pure b128 copies (no per-step transpose scatter).
// ---------------------------------------------------------------------------
__device__ __forceinline__ void gemm_mainloop(const __bf16* __restrict__ A,
                                              const __bf16* __restrict__ BT,
                                              int lda, int ldbt,
                                              int m0, int n0, int K,
                                              __bf16 (*As)[40], __bf16 (*Bs)[40],
                                              v8f acc[2][4]) {
  int tid   = threadIdx.x;
  int wave  = tid >> 5;
  int waveM = wave & 3;    // 0..3 -> 32-row band
  int waveN = wave >> 2;   // 0..1 -> 64-col band
  int r  = tid >> 2;           // 0..63 (+64 on second pass)
  int c8 = (tid & 3) * 8;      // 0,8,16,24
  for (int k0 = 0; k0 < K; k0 += 32) {
    // Stage A tile 128x32 and B^T tile 128x32: 4 b128 copies per thread
#pragma unroll
    for (int i = 0; i < 2; ++i) {
      int rr = r + i * 64;
      *(bf16x8*)&As[rr][c8] =
          *(const bf16x8*)(A + (size_t)(m0 + rr) * lda + k0 + c8);
      *(bf16x8*)&Bs[rr][c8] =
          *(const bf16x8*)(BT + (size_t)(n0 + rr) * ldbt + k0 + c8);
    }
    __syncthreads();
    bf16x16 af[2], bfr[4];
#pragma unroll
    for (int s = 0; s < 2; ++s)
      af[s] = load_frag_rm(&As[waveM * 32 + s * 16][0], 40);
#pragma unroll
    for (int s = 0; s < 4; ++s)
      bfr[s] = load_frag_rm(&Bs[waveN * 64 + s * 16][0], 40);
#pragma unroll
    for (int sm = 0; sm < 2; ++sm)
#pragma unroll
      for (int sn = 0; sn < 4; ++sn)
        acc[sm][sn] = wmma_bf16(af[sm], bfr[sn], acc[sm][sn]);
    __syncthreads();
  }
}

// ---------------------------------------------------------------------------
// Kernel 3: QKV GEMM + bias, scattering into Q [bh][t][64], K [bh][t][64],
// and V transposed [bh][64][t]. Subtile decomposition is wave-uniform:
// readfirstlane forces it onto the SALU so the q/k/v split is a scalar branch.
// ---------------------------------------------------------------------------
__global__ __launch_bounds__(256) void qkv_kernel(const __bf16* __restrict__ xn,
                                                  const __bf16* __restrict__ wqkvT,
                                                  const float* __restrict__ bqkv,
                                                  __bf16* __restrict__ qbuf,
                                                  __bf16* __restrict__ kbuf,
                                                  __bf16* __restrict__ vbuf) {
  __shared__ __bf16 As[128][40];
  __shared__ __bf16 Bs[128][40];
  v8f acc[2][4] = {};
  int m0 = blockIdx.y * 128;
  int n0 = blockIdx.x * 128;
  gemm_mainloop(xn, wqkvT, DD, DD, m0, n0, DD, As, Bs, acc);

  int lane = threadIdx.x & 31, half = lane >> 4, ln = lane & 15;
  int wave = threadIdx.x >> 5, waveM = wave & 3, waveN = wave >> 2;
#pragma unroll
  for (int sm = 0; sm < 2; ++sm) {
    int mbase = __builtin_amdgcn_readfirstlane(m0 + waveM * 32 + sm * 16);
    int b     = mbase >> 11;                      // band never crosses T bound
    int tbase = mbase & (TT - 1);
#pragma unroll
    for (int sn = 0; sn < 4; ++sn) {
      int nbase = __builtin_amdgcn_readfirstlane(n0 + waveN * 64 + sn * 16);
      int which = nbase >> 10;                    // 0=q 1=k 2=v (scalar)
      int dcol  = nbase & (DD - 1);
      int h     = dcol >> 6;
      int hdb   = dcol & (HD - 1);                // band stays inside one head
      int bh    = b * HH + h;
      float bias = bqkv[nbase + ln];
      v8f a = acc[sm][sn];
      if (which < 2) {
        __bf16* dst = (which ? kbuf : qbuf) +
                      ((size_t)bh * TT + tbase + 8 * half) * HD + hdb + ln;
#pragma unroll
        for (int r = 0; r < 8; ++r)
          dst[(size_t)r * HD] = (__bf16)(a[r] + bias);
      } else {
        __bf16* dst = vbuf + ((size_t)bh * HD + hdb + ln) * TT + tbase + 8 * half;
#pragma unroll
        for (int r = 0; r < 8; ++r)
          dst[r] = (__bf16)(a[r] + bias);         // transposed layout
      }
    }
  }
}

// ---------------------------------------------------------------------------
// Kernel 4: flash-style attention. 128 threads = 4 waves; each wave owns one
// 16-query tile, streams keys in chunks of 32 with online softmax.
//   S = Q(16x64) @ K^T        -> 2x wmma per 16-key subtile (d split 32+32)
//   P routed C-layout -> A-layout via per-wave LDS tile
//   O += P(16x32) @ V(32x64)  -> 4x wmma (V^T layout gives contiguous loads)
// ---------------------------------------------------------------------------
__global__ __launch_bounds__(128) void attn_kernel(const __bf16* __restrict__ q,
                                                   const __bf16* __restrict__ k,
                                                   const __bf16* __restrict__ vT,
                                                   __bf16* __restrict__ attnb) {
  __shared__ __bf16 Pst[4][16][40];
  int wave = threadIdx.x >> 5;
  int lane = threadIdx.x & 31;
  int half = lane >> 4, ln = lane & 15;
  int g  = blockIdx.x * 4 + wave;     // global q-tile id, 0..8191
  int bh = g >> 7;                    // 0..63
  int qi = g & 127;
  int t0 = qi * 16;
  int b = bh >> 4, h = bh & 15;

  const __bf16* Qb  = q  + (size_t)bh * TT * HD + (size_t)t0 * HD;
  const __bf16* Kb0 = k  + (size_t)bh * TT * HD;
  const __bf16* Vb0 = vT + (size_t)bh * HD * TT;

  bf16x16 qf0 = load_frag_rm(Qb, HD);
  bf16x16 qf1 = load_frag_rm(Qb + 32, HD);

  v8f o[4] = {};
  float rmax[8], rsum[8];
#pragma unroll
  for (int r = 0; r < 8; ++r) { rmax[r] = -3.0e38f; rsum[r] = 0.0f; }
  const float scale = 0.125f;   // HD^-0.5

  for (int kt = 0; kt < TT; kt += 32) {
    // --- scores for 32 keys (two 16-key subtiles) ---
    v8f s0 = {}, s1 = {};
    {
      const __bf16* Kb = Kb0 + (size_t)kt * HD;
      s0 = wmma_bf16(qf0, load_frag_rm(Kb, HD), s0);
      s0 = wmma_bf16(qf1, load_frag_rm(Kb + 32, HD), s0);
      const __bf16* Kc = Kb + 16 * HD;
      s1 = wmma_bf16(qf0, load_frag_rm(Kc, HD), s1);
      s1 = wmma_bf16(qf1, load_frag_rm(Kc + 32, HD), s1);
    }
    // --- online softmax (row = r + 8*half, cols across 16 lanes) ---
    float fac[8];
#pragma unroll
    for (int r = 0; r < 8; ++r) {
      float a0 = s0[r] * scale;
      float a1 = s1[r] * scale;
      float mx = fmaxf(a0, a1);
#pragma unroll
      for (int m = 8; m >= 1; m >>= 1) mx = fmaxf(mx, __shfl_xor(mx, m, 32));
      float mnew = fmaxf(rmax[r], mx);
      float f  = __expf(rmax[r] - mnew);
      float p0 = __expf(a0 - mnew);
      float p1 = __expf(a1 - mnew);
      float ad = p0 + p1;
#pragma unroll
      for (int m = 8; m >= 1; m >>= 1) ad += __shfl_xor(ad, m, 32);
      rsum[r] = rsum[r] * f + ad;
      rmax[r] = mnew;
      fac[r]  = f;
      int mrow = r + 8 * half;
      Pst[wave][mrow][ln]      = (__bf16)p0;
      Pst[wave][mrow][16 + ln] = (__bf16)p1;
    }
    // intra-wave LDS turnaround (DS ops are in-order within a wave; the wait +
    // memory clobber also pins the compiler's ordering)
    asm volatile("s_wait_dscnt 0" ::: "memory");
    bf16x16 pf = load_frag_rm(&Pst[wave][0][0], 40);
    // --- O = O*fac + P @ V ---
#pragma unroll
    for (int ns = 0; ns < 4; ++ns) {
#pragma unroll
      for (int r = 0; r < 8; ++r) o[ns][r] *= fac[r];
      bf16x16 vf = load_frag_rm(Vb0 + (size_t)(ns * 16) * TT + kt, TT);
      o[ns] = wmma_bf16(pf, vf, o[ns]);
    }
  }
  // --- normalize + write [B][T][D] bf16 ---
#pragma unroll
  for (int ns = 0; ns < 4; ++ns)
#pragma unroll
    for (int r = 0; r < 8; ++r) {
      int t   = t0 + r + 8 * half;
      int col = h * HD + ns * 16 + ln;
      attnb[((size_t)b * TT + t) * DD + col] = (__bf16)(o[ns][r] / rsum[r]);
    }
}

// ---------------------------------------------------------------------------
// Kernel 5: output projection + bias + residual (fp32 out)
// ---------------------------------------------------------------------------
__global__ __launch_bounds__(256) void proj_kernel(const __bf16* __restrict__ attnb,
                                                   const __bf16* __restrict__ wprojT,
                                                   const float* __restrict__ bproj,
                                                   const float* __restrict__ x,
                                                   float* __restrict__ out) {
  __shared__ __bf16 As[128][40];
  __shared__ __bf16 Bs[128][40];
  v8f acc[2][4] = {};
  int m0 = blockIdx.y * 128;
  int n0 = blockIdx.x * 128;
  gemm_mainloop(attnb, wprojT, DD, DD, m0, n0, DD, As, Bs, acc);

  int lane = threadIdx.x & 31, half = lane >> 4, ln = lane & 15;
  int wave = threadIdx.x >> 5, waveM = wave & 3, waveN = wave >> 2;
#pragma unroll
  for (int sm = 0; sm < 2; ++sm) {
    int mbase = m0 + waveM * 32 + sm * 16 + 8 * half;
#pragma unroll
    for (int sn = 0; sn < 4; ++sn) {
      int nbase = n0 + waveN * 64 + sn * 16;
      float bias = bproj[nbase + ln];
      v8f a = acc[sm][sn];
      const float* xs = x   + (size_t)mbase * DD + nbase + ln;
      float*       os = out + (size_t)mbase * DD + nbase + ln;
#pragma unroll
      for (int r = 0; r < 8; ++r)
        os[(size_t)r * DD] = xs[(size_t)r * DD] + a[r] + bias;
    }
  }
}

// ---------------------------------------------------------------------------
// Launcher
// ---------------------------------------------------------------------------
extern "C" void kernel_launch(void* const* d_in, const int* in_sizes, int n_in,
                              void* d_out, int out_size, void* d_ws, size_t ws_size,
                              hipStream_t stream) {
  const float* x      = (const float*)d_in[0];
  const float* w_qkv  = (const float*)d_in[1];
  const float* b_qkv  = (const float*)d_in[2];
  const float* w_proj = (const float*)d_in[3];
  const float* b_proj = (const float*)d_in[4];
  const float* ln_g   = (const float*)d_in[5];
  const float* ln_b   = (const float*)d_in[6];
  float* out = (float*)d_out;

  // Workspace carve-up (bytes)
  char* ws = (char*)d_ws;
  size_t off = 0;
  __bf16* xn    = (__bf16*)(ws + off); off += (size_t)MROWS * DD * 2;        // 16 MiB (reused as attn out)
  __bf16* wq_bT = (__bf16*)(ws + off); off += (size_t)DD * 3 * DD * 2;       //  6 MiB  [3072][1024]
  __bf16* wp_bT = (__bf16*)(ws + off); off += (size_t)DD * DD * 2;           //  2 MiB  [1024][1024]
  __bf16* qbuf  = (__bf16*)(ws + off); off += (size_t)BB * HH * TT * HD * 2; // 16 MiB
  __bf16* kbuf  = (__bf16*)(ws + off); off += (size_t)BB * HH * TT * HD * 2; // 16 MiB
  __bf16* vbuf  = (__bf16*)(ws + off); off += (size_t)BB * HH * TT * HD * 2; // 16 MiB
  __bf16* attnb = xn;  // xn is dead after the QKV GEMM; reuse for attn output

  // 1) LayerNorm -> bf16
  ln_kernel<<<MROWS, 256, 0, stream>>>(x, ln_g, ln_b, xn);

  // 2) Weight cast + transpose -> bf16 W^T
  {
    dim3 g1(3 * DD / 32, DD / 32);         // 96 x 32 : wqkv [1024][3072] -> [3072][1024]
    castT_kernel<<<g1, 256, 0, stream>>>(w_qkv, wq_bT, DD, 3 * DD);
    dim3 g2(DD / 32, DD / 32);             // 32 x 32 : wproj
    castT_kernel<<<g2, 256, 0, stream>>>(w_proj, wp_bT, DD, DD);
  }

  // 3) QKV GEMM + bias + head scatter (V transposed)
  {
    dim3 grid(3 * DD / 128, MROWS / 128);  // 24 x 64
    qkv_kernel<<<grid, 256, 0, stream>>>(xn, wq_bT, b_qkv, qbuf, kbuf, vbuf);
  }

  // 4) Attention (flash-style, online softmax)
  {
    int qtiles = BB * HH * (TT / 16);      // 8192 wave-tiles, 4 per block
    attn_kernel<<<qtiles / 4, 128, 0, stream>>>(qbuf, kbuf, vbuf, attnb);
  }

  // 5) Output projection + bias + residual
  {
    dim3 grid(DD / 128, MROWS / 128);      // 8 x 64
    proj_kernel<<<grid, 256, 0, stream>>>(attnb, wp_bT, b_proj, x, out);
  }
}
